// GRU_85392539779945
// MI455X (gfx1250) — compile-verified, matrix operand order checked by reading, not run on hardware
//
#include <hip/hip_runtime.h>
#include <hip/hip_bf16.h>

// Problem sizes (fixed by reference)
#define NB   128
#define LSEQ 512
#define HD   512
#define GD   1536   // 3*H

typedef __attribute__((ext_vector_type(16))) _Float16 v16h;
typedef __attribute__((ext_vector_type(8)))  _Float16 f16x8;
typedef __attribute__((ext_vector_type(8)))  float    v8f;
typedef __attribute__((ext_vector_type(8)))  float    f32x8;

// TDM descriptor argument types (6-arg builtin signature on this toolchain)
typedef __attribute__((ext_vector_type(4))) unsigned int u32x4;
typedef __attribute__((ext_vector_type(8))) int          i32x8;
typedef __attribute__((ext_vector_type(4))) int          i32x4;

__device__ __forceinline__ float sigmoidf_(float x) {
    return 1.0f / (1.0f + __expf(-x));
}

// Assemble a 16x16x32-f16 WMMA B fragment for this lane.
// Per ISA: lanes 0-15 hold K = k0..k0+7 and k0+16..k0+23 for column N=lane;
// lanes 16-31 hold K = k0+8.. and k0+24.. (khalf folded into k0 by caller).
__device__ __forceinline__ v16h load_b_frag(const _Float16* __restrict__ p, int k0) {
    f16x8 lo = *(const f16x8*)(p + k0);
    f16x8 hi = *(const f16x8*)(p + k0 + 16);
    v16h b;
#pragma unroll
    for (int i = 0; i < 8; ++i) { b[i] = lo[i]; b[8 + i] = hi[i]; }
    return b;
}

// ---------------------------------------------------------------------------
// Kernel 0: f32 -> f16 weight conversion (W_ih then W_hh)
// ---------------------------------------------------------------------------
__global__ void cvt_weights(const float* __restrict__ wih, const float* __restrict__ whh,
                            _Float16* __restrict__ wih16, _Float16* __restrict__ whh16) {
    int i = blockIdx.x * blockDim.x + threadIdx.x;
    if (i < GD * HD) {
        wih16[i] = (_Float16)wih[i];
    } else {
        int j = i - GD * HD;
        if (j < GD * HD) whh16[j] = (_Float16)whh[j];
    }
}

// ---------------------------------------------------------------------------
// Kernel 1: gx = input @ W_ih^T + b_ih   ([65536 x 512] * [512 x 1536])
// Block = 256 threads (8 waves). Block tile 64(M) x 128(N).
// Wave (mi, ni): 16 rows x 64 cols = 4 WMMA accumulators, K-loop over 512.
// ---------------------------------------------------------------------------
__global__ __launch_bounds__(256)
void gx_gemm(const float* __restrict__ x,        // [N*L, H]
             const _Float16* __restrict__ w16,   // [G, H]
             const float* __restrict__ bih,      // [G]
             _Float16* __restrict__ gx)          // [N*L, G]
{
    const int lane  = threadIdx.x & 31;
    const int wave  = threadIdx.x >> 5;
    const int mi    = wave & 3;                  // 0..3
    const int ni    = wave >> 2;                 // 0..1
    const int mBase = blockIdx.y * 64 + mi * 16;
    const int nBase = blockIdx.x * 128 + ni * 64;

    const int lrow  = lane & 15;
    const int khalf = (lane >> 4) * 8;           // 0 or 8
    const int hiRow = (lane >> 4) * 8;           // C-layout row offset

    v8f acc[4] = {};

    const float* aRow = x + (size_t)(mBase + lrow) * HD;
    const _Float16* bp[4];
#pragma unroll
    for (int j = 0; j < 4; ++j)
        bp[j] = w16 + (size_t)(nBase + j * 16 + lrow) * HD;

    for (int kt = 0; kt < HD; kt += 32) {
        const int k0 = kt + khalf;
        __builtin_prefetch(aRow + kt + 128, 0, 3);   // global_prefetch_b8 (near)
        f32x8 alo = *(const f32x8*)(aRow + k0);
        f32x8 ahi = *(const f32x8*)(aRow + k0 + 16);
        v16h a;
#pragma unroll
        for (int i = 0; i < 8; ++i) { a[i] = (_Float16)alo[i]; a[8 + i] = (_Float16)ahi[i]; }
#pragma unroll
        for (int j = 0; j < 4; ++j) {
            v16h b = load_b_frag(bp[j], k0);
            acc[j] = __builtin_amdgcn_wmma_f32_16x16x32_f16(
                false, a, false, b, (short)0, acc[j], false, false);
        }
    }

#pragma unroll
    for (int j = 0; j < 4; ++j) {
        const int col  = nBase + j * 16 + lrow;
        const float bs = bih[col];
#pragma unroll
        for (int v = 0; v < 8; ++v) {
            const int r = mBase + v + hiRow;
            gx[(size_t)r * GD + col] = (_Float16)(acc[j][v] + bs);
        }
    }
}

// ---------------------------------------------------------------------------
// Kernel 2: sequential GRU scan. 8 blocks (16 batch rows each), 512 threads
// (16 waves). h lives in LDS (f32 master + masked f16 WMMA-A shadow).
// Wave w owns output columns [32w, 32w+32) of each of the 3 gates:
// 6 WMMA N-tiles -> gate fusion is wave-local.
// The per-step gx gate tile [16 x 1536] f16 is DMA'd into LDS by the Tensor
// Data Mover (issued before the GEMM, waited after it -> fully overlapped).
// ---------------------------------------------------------------------------
__global__ __launch_bounds__(512)
void gru_scan(const _Float16* __restrict__ gx,     // [N*L, G]
              const _Float16* __restrict__ whh16,  // [G, H]
              const float* __restrict__ h0,        // [N, H]
              const int*   __restrict__ is_init,   // [N*L]
              const float* __restrict__ bhh,       // [G]
              float* __restrict__ seq,             // [N*L, H]
              float* __restrict__ hlast)           // [N, H]
{
    __shared__ float    hS[16][HD];      // 32 KB, f32 master hidden state
    __shared__ _Float16 hcS[16][HD];     // 16 KB, masked f16 copy (WMMA A)
    __shared__ _Float16 gxS[16][GD];     // 48 KB, per-step gx tile (TDM dest)
    __shared__ float    mS[16];          // per-row episode mask

    const int lane  = threadIdx.x & 31;
    const int wave  = threadIdx.x >> 5;  // 0..15
    const int nBase = blockIdx.x * 16;   // batch-row tile
    const int lrow  = lane & 15;
    const int khalf = (lane >> 4) * 8;
    const int hiRow = (lane >> 4) * 8;

    // load h0 tile
    for (int idx = threadIdx.x; idx < 16 * HD; idx += 512) {
        int r = idx >> 9, c = idx & (HD - 1);
        hS[r][c] = h0[(size_t)(nBase + r) * HD + c];
    }

    // per-wave W_hh row pointers + b_hh for the 6 owned N-tiles
    const int c0 = wave * 32;            // column base within one gate
    const _Float16* bp[6];
    float bb[6];
#pragma unroll
    for (int g = 0; g < 3; ++g)
#pragma unroll
        for (int s = 0; s < 2; ++s) {
            int col = g * HD + c0 + s * 16 + lrow;
            bp[2 * g + s] = whh16 + (size_t)col * HD;
            bb[2 * g + s] = bhh[col];
        }

    // Invariant parts of the TDM D# for the gx tile:
    // 2D tensor, data_size = 2B, tile = 1536(x) x 16(y),
    // tensor_dim0 = 1536, tensor_dim1 = 16, dim0_stride = L*3H = 786432.
    const unsigned ldsAddr = (unsigned)(size_t)&gxS[0][0];   // low 32 bits = LDS offset
    i32x8 dg1;
    dg1[0] = 0x00010000;                 // workgroup_mask=0, data_size=1 (2 bytes)
    dg1[1] = (GD & 0xFFFF) << 16;        // tensor_dim0[15:0]
    dg1[2] = (GD >> 16) | (16 << 16);    // tensor_dim0[31:16], tensor_dim1[15:0]
    dg1[3] = (GD & 0xFFFF) << 16;        // tensor_dim1[31:16]=0, tile_dim0=1536
    dg1[4] = 16;                         // tile_dim1=16, tile_dim2=0
    dg1[5] = LSEQ * GD;                  // tensor_dim0_stride[31:0] (elements)
    dg1[6] = 0;                          // stride[47:32]=0, dim1_stride[15:0]=0
    dg1[7] = 0;
    const i32x4 dz4 = {0, 0, 0, 0};      // groups 2/3 disabled (2D tensor)
    const i32x8 dz8 = {0, 0, 0, 0, 0, 0, 0, 0};

    for (int t = 0; t < LSEQ; ++t) {
        __syncthreads();                 // prior combine done: h stable, gxS free

        // One wave kicks off the TDM load of this step's gx tile (EXEC-independent
        // DMA; overlaps the hc copy + the whole recurrent GEMM below).
        if (wave == 0) {
            unsigned long long ga =
                (unsigned long long)(size_t)(gx + ((size_t)nBase * LSEQ + t) * GD);
            u32x4 dg0;
            dg0[0] = 1u;                                              // count=1, user mode
            dg0[1] = ldsAddr;                                         // lds_addr
            dg0[2] = (unsigned)(ga & 0xFFFFFFFFu);                    // global_addr[31:0]
            dg0[3] = (unsigned)((ga >> 32) & 0x1FFFFFFu) | 0x80000000u; // addr[56:32], type=2
            __builtin_amdgcn_tensor_load_to_lds(dg0, dg1, dz4, dz4, dz8, 0);
        }

        // masked f16 shadow: hc = h * (1 - is_initial[:, t])
        for (int idx = threadIdx.x; idx < 16 * HD; idx += 512) {
            int r = idx >> 9, c = idx & (HD - 1);
            float m = 1.0f - (float)is_init[(size_t)(nBase + r) * LSEQ + t];
            if (c == 0) mS[r] = m;
            hcS[r][c] = (_Float16)(hS[r][c] * m);
        }
        __syncthreads();

        // gh = hc @ W_hh^T for this wave's 6 tiles
        v8f acc[6] = {};
        for (int kt = 0; kt < HD; kt += 32) {
            const int k0 = kt + khalf;
            f16x8 alo = *(const f16x8*)&hcS[lrow][k0];
            f16x8 ahi = *(const f16x8*)&hcS[lrow][k0 + 16];
            v16h a;
#pragma unroll
            for (int i = 0; i < 8; ++i) { a[i] = alo[i]; a[8 + i] = ahi[i]; }
#pragma unroll
            for (int j = 0; j < 6; ++j) {
                v16h b = load_b_frag(bp[j], k0);
                acc[j] = __builtin_amdgcn_wmma_f32_16x16x32_f16(
                    false, a, false, b, (short)0, acc[j], false, false);
            }
        }

        // make the DMA'd gx tile visible to every wave
        if (wave == 0) __builtin_amdgcn_s_wait_tensorcnt(0);
        __syncthreads();

        // fuse gates + state update (wave-local columns)
#pragma unroll
        for (int s = 0; s < 2; ++s) {
            const int col = c0 + s * 16 + lrow;        // in [0, 512)
#pragma unroll
            for (int v = 0; v < 8; ++v) {
                const int r = v + hiRow;               // 0..15
                const size_t m = (size_t)(nBase + r) * LSEQ + t;
                float gr  = (float)gxS[r][col]          + acc[s][v]     + bb[s];
                float gz  = (float)gxS[r][col + HD]     + acc[2 + s][v] + bb[2 + s];
                float gn  = (float)gxS[r][col + 2 * HD];
                float ghn = acc[4 + s][v] + bb[4 + s];
                float rr  = sigmoidf_(gr);
                float zz  = sigmoidf_(gz);
                float nn  = tanhf(gn + rr * ghn);
                float hc  = hS[r][col] * mS[r];
                float hn  = (1.0f - zz) * nn + zz * hc;
                seq[m * HD + col] = hn;
                hS[r][col] = hn;
            }
        }
    }

    __syncthreads();
    for (int idx = threadIdx.x; idx < 16 * HD; idx += 512) {
        int r = idx >> 9, c = idx & (HD - 1);
        hlast[(size_t)(nBase + r) * HD + c] = hS[r][c];
    }
}

// ---------------------------------------------------------------------------
// Kernel 3: residual + LayerNorm + h_last broadcast. One block per row.
// ---------------------------------------------------------------------------
__global__ __launch_bounds__(256)
void ln_out(const float* __restrict__ x,      // input [N*L, H]
            const float* __restrict__ seq,    // [N*L, H]
            const float* __restrict__ hlast,  // [N, H]
            const float* __restrict__ gamma,
            const float* __restrict__ beta,
            float* __restrict__ out,          // [N*L, H]
            float* __restrict__ out2)         // [N*L, H]
{
    __shared__ float sS[256], sQ[256];
    const int row = blockIdx.x;
    const size_t base = (size_t)row * HD;
    const int tid = threadIdx.x;

    float y0 = seq[base + tid]       + x[base + tid];
    float y1 = seq[base + tid + 256] + x[base + tid + 256];
    sS[tid] = y0 + y1;
    sQ[tid] = y0 * y0 + y1 * y1;
    __syncthreads();
#pragma unroll
    for (int off = 128; off > 0; off >>= 1) {
        if (tid < off) { sS[tid] += sS[tid + off]; sQ[tid] += sQ[tid + off]; }
        __syncthreads();
    }
    const float mu   = sS[0] * (1.0f / HD);
    const float var  = sQ[0] * (1.0f / HD) - mu * mu;
    const float rstd = rsqrtf(var + 1e-5f);

    out[base + tid]       = (y0 - mu) * rstd * gamma[tid]       + beta[tid];
    out[base + tid + 256] = (y1 - mu) * rstd * gamma[tid + 256] + beta[tid + 256];

    const int n = row >> 9;          // row = n*L + l, L = 512
    out2[base + tid]       = hlast[(size_t)n * HD + tid];
    out2[base + tid + 256] = hlast[(size_t)n * HD + tid + 256];
}

// ---------------------------------------------------------------------------
extern "C" void kernel_launch(void* const* d_in, const int* in_sizes, int n_in,
                              void* d_out, int out_size, void* d_ws, size_t ws_size,
                              hipStream_t stream) {
    const float* x     = (const float*)d_in[0];   // input  [N,L,H]
    const float* h0    = (const float*)d_in[1];   // h      [N,H]
    const int*   iinit = (const int*)  d_in[2];   // is_initial [N,L,1]
    const float* wih   = (const float*)d_in[3];   // W_ih   [3H,H]
    const float* whh   = (const float*)d_in[4];   // W_hh   [3H,H]
    const float* bih   = (const float*)d_in[5];   // b_ih   [3H]
    const float* bhh   = (const float*)d_in[6];   // b_hh   [3H]
    const float* gamma = (const float*)d_in[7];   // [H]
    const float* beta  = (const float*)d_in[8];   // [H]

    float* out  = (float*)d_out;                          // output [N,L,H]
    float* out2 = out + (size_t)NB * LSEQ * HD;           // h_exp  [N,L,H]

    char* ws = (char*)d_ws;
    _Float16* wih16 = (_Float16*)ws;  ws += (size_t)GD * HD * sizeof(_Float16);
    _Float16* whh16 = (_Float16*)ws;  ws += (size_t)GD * HD * sizeof(_Float16);
    _Float16* gxbuf = (_Float16*)ws;  ws += (size_t)NB * LSEQ * GD * sizeof(_Float16);
    float*    seq   = (float*)ws;     ws += (size_t)NB * LSEQ * HD * sizeof(float);
    float*    hlast = (float*)ws;

    cvt_weights<<<(2 * GD * HD + 255) / 256, 256, 0, stream>>>(wih, whh, wih16, whh16);
    gx_gemm<<<dim3(GD / 128, (NB * LSEQ) / 64), 256, 0, stream>>>(x, wih16, bih, gxbuf);
    gru_scan<<<NB / 16, 512, 0, stream>>>(gxbuf, whh16, h0, iinit, bhh, seq, hlast);
    ln_out<<<NB * LSEQ, 256, 0, stream>>>(x, seq, hlast, gamma, beta, out, out2);
}
